// Model_21260088115221
// MI455X (gfx1250) — matrix operand order repaired and checked by reading, not compiled, not run on hardware
//
#include <hip/hip_runtime.h>
#include <hip/hip_bf16.h>

// ---------------------------------------------------------------------------
// Model constants
// ---------------------------------------------------------------------------
#define BSZ     16
#define SEQLEN  512
#define PREDLEN 1024
#define C_IN    32
#define C_OUT   32
#define DMODEL  512
#define DSTATE  32
#define DCONV   4
#define DINNER  1024      // 2 * DMODEL
#define DTRANK  32
#define NMARK   4
#define MROWS   (BSZ * SEQLEN)        // 8192 rows per mamba iteration
#define CATROWS (BSZ * PREDLEN)       // 16384 rows for head GEMM

// Set to 0 to fall back to synchronous global->reg->LDS staging.
#define USE_ASYNC_COPY 1

// ---------------------------------------------------------------------------
// WMMA vector types (wave32, CDNA5)
// ---------------------------------------------------------------------------
typedef __attribute__((ext_vector_type(16))) __bf16 v16bf;
typedef __attribute__((ext_vector_type(8)))  __bf16 v8bf;
typedef __attribute__((ext_vector_type(8)))  float  v8f;

union BFFrag { v16bf v; v8bf h[2]; };

__device__ __forceinline__ unsigned short f2bf(float f) {
    unsigned u = __float_as_uint(f);
    unsigned rounding = 0x7FFFu + ((u >> 16) & 1u);
    return (unsigned short)((u + rounding) >> 16);
}

// ---------------------------------------------------------------------------
// 32B (2 x b128) global -> LDS copy.
// Async path: CDNA5 GLOBAL_LOAD_ASYNC_TO_LDS_B128 (tracked by ASYNCcnt).
// The LDS destination address is the low 32 bits of the generic pointer
// ---------------------------------------------------------------------------
__device__ __forceinline__ void stage_copy32(const unsigned short* __restrict__ g,
                                             unsigned short* lds)
{
#if USE_ASYNC_COPY
    unsigned       lo = (unsigned)(uintptr_t)lds;
    unsigned long long ga = (unsigned long long)(uintptr_t)g;
    asm volatile("global_load_async_to_lds_b128 %0, %1, off"
                 :: "v"(lo), "v"(ga) : "memory");
    asm volatile("global_load_async_to_lds_b128 %0, %1, off"
                 :: "v"(lo + 16u), "v"(ga + 16ull) : "memory");
#else
    const uint4* s = (const uint4*)g;
    uint4* d = (uint4*)lds;
    d[0] = s[0];
    d[1] = s[1];
#endif
}

__device__ __forceinline__ void wait_async0()
{
#if USE_ASYNC_COPY
    asm volatile("s_wait_asynccnt 0x0" ::: "memory");
#endif
}

// ---------------------------------------------------------------------------
// Generic bf16 WMMA GEMM:  C[M,N] (f32) = A[M,K] (bf16, row-major, stride lda)
//                                       x Bw[N,K]^T (bf16, row-major, stride ldb)
// Block: 256 threads = 8 waves as 4(M) x 2(N). Tile 128(M) x 128(N), K step 32.
// Each wave: 2 M-frags x 4 N-frags = 8 WMMAs per K-step.
// LDS double-buffered; rows padded to 40 ushorts (80B) -> conflict-free b128.
// Requires: M % 128 == 0, K % 32 == 0 (true at all call sites); N guarded
// (OOB B rows clamped -- their accumulator columns are never stored).
// ---------------------------------------------------------------------------
#define TLM 128
#define TLN 128
#define LDSROW 40

__global__ __launch_bounds__(256)
void gemm_bf16_wmma(const unsigned short* __restrict__ A,
                    const unsigned short* __restrict__ Bw,
                    float* __restrict__ C,
                    int M, int N, int K, int lda, int ldb, int ldc)
{
    __shared__ __align__(16) unsigned short As[2][TLM * LDSROW];
    __shared__ __align__(16) unsigned short Bs[2][TLN * LDSROW];

    const int tid  = threadIdx.x;
    const int wave = tid >> 5;
    const int lane = tid & 31;
    const int wm   = wave >> 1;   // 0..3 : 32-row M slice
    const int wn   = wave & 1;    // 0..1 : 64-col N slice
    const int mt   = blockIdx.y * TLM;
    const int nt   = blockIdx.x * TLN;

    // cooperative-load mapping: 256 threads = 128 rows x 2 half-rows (32B)
    const int lr = tid >> 1;
    const int lh = (tid & 1) * 16;            // ushort offset of half-row
    const int arow = mt + lr;
    int brow = nt + lr;
    if (brow > N - 1) brow = N - 1;           // clamp: OOB cols discarded at store

    v8f acc[2][4] = {};

    auto issue_stage = [&](int k0, int buf) {
        stage_copy32(A  + (size_t)arow * lda + k0 + lh, &As[buf][lr * LDSROW + lh]);
        stage_copy32(Bw + (size_t)brow * ldb + k0 + lh, &Bs[buf][lr * LDSROW + lh]);
    };

    issue_stage(0, 0);
    wait_async0();
    __syncthreads();

    int buf = 0;
    for (int k0 = 0; k0 < K; k0 += 32) {
        const bool more = (k0 + 32 < K);
        if (more) issue_stage(k0 + 32, buf ^ 1);   // overlap with WMMAs below

        // ---- A fragments (16x32 bf16): lanes 0-15 K{0..7,16..23}, 16-31 K{8..15,24..31}
        BFFrag af[2];
        #pragma unroll
        for (int mi = 0; mi < 2; ++mi) {
            const int row = wm * 32 + mi * 16 + (lane & 15);
            const int kb  = (lane < 16) ? 0 : 8;
            const unsigned short* p = &As[buf][row * LDSROW];
            af[mi].h[0] = *(const v8bf*)(p + kb);
            af[mi].h[1] = *(const v8bf*)(p + kb + 16);
        }
        // ---- B fragments (32x16 bf16): lanes 0-15 K{0..7,16..23}, 16-31 K{8..15,24..31}
        BFFrag bfr[4];
        #pragma unroll
        for (int ni = 0; ni < 4; ++ni) {
            const int row = wn * 64 + ni * 16 + (lane & 15);
            const int kb  = (lane < 16) ? 0 : 8;
            const unsigned short* p = &Bs[buf][row * LDSROW] + kb;
            bfr[ni].h[0] = *(const v8bf*)(p);
            bfr[ni].h[1] = *(const v8bf*)(p + 16);
        }
        // ---- 8 back-to-back WMMAs
        #pragma unroll
        for (int mi = 0; mi < 2; ++mi)
            #pragma unroll
            for (int ni = 0; ni < 4; ++ni)
                acc[mi][ni] = __builtin_amdgcn_wmma_f32_16x16x32_bf16(
                    false, af[mi].v, false, bfr[ni].v, (short)0, acc[mi][ni],
                    false, false);

        if (more) wait_async0();
        __syncthreads();
        buf ^= 1;
    }

    // ---- store D: lane(0-15)=col, VGPR v = row v (lanes<16) / v+8 (lanes>=16)
    #pragma unroll
    for (int mi = 0; mi < 2; ++mi) {
        const int gm0 = mt + wm * 32 + mi * 16 + ((lane < 16) ? 0 : 8);
        #pragma unroll
        for (int ni = 0; ni < 4; ++ni) {
            const int gn = nt + wn * 64 + ni * 16 + (lane & 15);
            if (gn >= N) continue;
            #pragma unroll
            for (int v = 0; v < 8; ++v) {
                const int gm = gm0 + v;
                if (gm < M) C[(size_t)gm * ldc + gn] = acc[mi][ni][v];
            }
        }
    }
}

// ---------------------------------------------------------------------------
// f32 -> bf16 bit conversion
// ---------------------------------------------------------------------------
__global__ void f32_to_bf16_kernel(const float* __restrict__ in,
                                   unsigned short* __restrict__ out, int n)
{
    int i = blockIdx.x * blockDim.x + threadIdx.x;
    if (i < n) out[i] = f2bf(in[i]);
}

// A = -exp(A_log)
__global__ void aneg_kernel(const float* __restrict__ Alog, float* __restrict__ An, int n)
{
    int i = blockIdx.x * blockDim.x + threadIdx.x;
    if (i < n) An[i] = -__expf(Alog[i]);
}

// ---------------------------------------------------------------------------
// Embedding: token conv (k=3, edge pad) + temporal proj + sinusoidal pos.
// ---------------------------------------------------------------------------
__global__ void embed_kernel(const float* __restrict__ x_enc,   // (16,512,32)
                             const float* __restrict__ cw,      // (512,32,3)
                             const float* __restrict__ xm,      // (16,512,4)
                             const float* __restrict__ tw,      // (512,4)
                             unsigned short* __restrict__ cur_bf)
{
    int idx = blockIdx.x * blockDim.x + threadIdx.x;
    if (idx >= MROWS * DMODEL) return;
    const int d   = idx & (DMODEL - 1);
    const int row = idx >> 9;           // b*512 + l
    const int b   = row >> 9;
    const int l   = row & (SEQLEN - 1);

    float acc = 0.f;
    #pragma unroll
    for (int t = 0; t < 3; ++t) {
        int lc = l + t - 2;
        lc = lc < 0 ? 0 : lc;
        const float* xp = x_enc + (size_t)(b * SEQLEN + lc) * C_IN;
        const float* wp = cw + (size_t)d * (C_IN * 3) + t;
        #pragma unroll
        for (int c = 0; c < C_IN; ++c) acc += xp[c] * wp[c * 3];
    }
    const float* mp = xm + (size_t)row * NMARK;
    const float* tp = tw + (size_t)d * NMARK;
    #pragma unroll
    for (int m = 0; m < NMARK; ++m) acc += mp[m] * tp[m];

    const int j = d >> 1;
    const float div = __expf((float)(2 * j) * (-9.210340371976184f / (float)DMODEL));
    const float ang = (float)l * div;
    acc += (d & 1) ? __cosf(ang) : __sinf(ang);

    cur_bf[idx] = f2bf(acc);
}

// ---------------------------------------------------------------------------
// Causal depthwise conv (k=4) + bias + SiLU over the x half of xz.
// ---------------------------------------------------------------------------
__global__ void conv_silu_kernel(const float* __restrict__ xz,    // (8192, 2048)
                                 const float* __restrict__ cw,    // (1024,1,4)
                                 const float* __restrict__ cb,    // (1024)
                                 float* __restrict__ xact,
                                 unsigned short* __restrict__ xact_bf)
{
    int idx = blockIdx.x * blockDim.x + threadIdx.x;
    if (idx >= MROWS * DINNER) return;
    const int d   = idx & (DINNER - 1);
    const int row = idx >> 10;
    const int l   = row & (SEQLEN - 1);

    float acc = cb[d];
    #pragma unroll
    for (int t = 0; t < DCONV; ++t) {
        const int lt = l + t - (DCONV - 1);
        if (lt >= 0)
            acc += xz[(size_t)(row + t - (DCONV - 1)) * (2 * DINNER) + d] * cw[d * DCONV + t];
    }
    const float sig = 1.f / (1.f + __expf(-acc));
    const float o = acc * sig;
    xact[idx] = o;
    xact_bf[idx] = f2bf(o);
}

// dt = softplus(dt_pre + bias)
__global__ void dt_softplus_kernel(const float* __restrict__ pre,
                                   const float* __restrict__ bias,
                                   float* __restrict__ dt)
{
    int idx = blockIdx.x * blockDim.x + threadIdx.x;
    if (idx >= MROWS * DINNER) return;
    const int d = idx & (DINNER - 1);
    const float v = pre[idx] + bias[d];
    dt[idx] = (v > 20.f) ? v : log1pf(__expf(v));
}

// ---------------------------------------------------------------------------
// Selective scan: one thread per (b, d) channel; h[32] in VGPRs; B_t/C_t in
// LDS. Fuses (y + x*D) * silu(z) gating; emits bf16 for the out_proj GEMM.
// ---------------------------------------------------------------------------
__global__ __launch_bounds__(256)
void scan_kernel(const float* __restrict__ dt,     // (8192,1024)
                 const float* __restrict__ xact,   // (8192,1024)
                 const float* __restrict__ xdbc,   // (8192,96): [dt | B | C]
                 const float* __restrict__ Aneg,   // (1024,32)
                 const float* __restrict__ Dv,     // (1024)
                 const float* __restrict__ xz,     // (8192,2048): z at cols 1024..
                 unsigned short* __restrict__ yg)  // (8192,1024) bf16
{
    const int b = blockIdx.x >> 2;
    const int d = (blockIdx.x & 3) * 256 + threadIdx.x;

    __shared__ float sB[DSTATE];
    __shared__ float sC[DSTATE];

    float Ar[DSTATE], h[DSTATE];
    #pragma unroll
    for (int s = 0; s < DSTATE; ++s) {
        Ar[s] = Aneg[(size_t)d * DSTATE + s];
        h[s]  = 0.f;
    }
    const float Dd = Dv[d];

    for (int l = 0; l < SEQLEN; ++l) {
        const int r = b * SEQLEN + l;
        if (threadIdx.x < 2 * DSTATE) {
            const float v = xdbc[(size_t)r * 96 + DTRANK + threadIdx.x];
            if (threadIdx.x < DSTATE) sB[threadIdx.x] = v;
            else                      sC[threadIdx.x - DSTATE] = v;
        }
        __syncthreads();

        const float dtv = dt[(size_t)r * DINNER + d];
        const float xv  = xact[(size_t)r * DINNER + d];
        const float dtx = dtv * xv;
        float y = 0.f;
        #pragma unroll
        for (int s = 0; s < DSTATE; ++s) {
            h[s] = __expf(dtv * Ar[s]) * h[s] + dtx * sB[s];
            y += h[s] * sC[s];
        }

        const float zv  = xz[(size_t)r * (2 * DINNER) + DINNER + d];
        const float sig = 1.f / (1.f + __expf(-zv));
        const float o   = (y + xv * Dd) * (zv * sig);
        yg[(size_t)r * DINNER + d] = f2bf(o);
        __syncthreads();
    }
}

// ---------------------------------------------------------------------------
// LayerNorm: one wave (32 lanes) per 512-elem row; wave32 shfl reductions.
// ---------------------------------------------------------------------------
__global__ __launch_bounds__(256)
void layernorm_kernel(const float* __restrict__ x,      // (8192,512)
                      const float* __restrict__ w,
                      const float* __restrict__ bias,
                      unsigned short* __restrict__ cur_bf,  // (8192,512)
                      unsigned short* __restrict__ cat_bf,  // (16384,512)
                      int it)
{
    const int row  = blockIdx.x * 8 + (threadIdx.x >> 5);
    const int lane = threadIdx.x & 31;
    const float* xr = x + (size_t)row * DMODEL;

    float vals[16];
    float s = 0.f, sq = 0.f;
    #pragma unroll
    for (int k = 0; k < 16; ++k) {
        const float v = xr[lane + 32 * k];
        vals[k] = v;
        s += v;
        sq += v * v;
    }
    #pragma unroll
    for (int m = 16; m >= 1; m >>= 1) {
        s  += __shfl_xor(s,  m, 32);
        sq += __shfl_xor(sq, m, 32);
    }
    const float mu  = s * (1.f / (float)DMODEL);
    const float var = sq * (1.f / (float)DMODEL) - mu * mu;
    const float rs  = rsqrtf(var + 1e-5f);

    const int b = row >> 9;
    const int l = row & (SEQLEN - 1);
    const size_t crow = ((size_t)b * PREDLEN + (size_t)it * SEQLEN + l) * DMODEL;

    #pragma unroll
    for (int k = 0; k < 16; ++k) {
        const int d = lane + 32 * k;
        const float o = (vals[k] - mu) * rs * w[d] + bias[d];
        const unsigned short ob = f2bf(o);
        cur_bf[(size_t)row * DMODEL + d] = ob;
        cat_bf[crow + d] = ob;
    }
}

// ---------------------------------------------------------------------------
// Launch
// ---------------------------------------------------------------------------
extern "C" void kernel_launch(void* const* d_in, const int* in_sizes, int n_in,
                              void* d_out, int out_size, void* d_ws, size_t ws_size,
                              hipStream_t stream)
{
    const float* x_enc      = (const float*)d_in[0];
    const float* x_mark_enc = (const float*)d_in[1];
    const float* conv_emb_w = (const float*)d_in[4];
    const float* temp_w     = (const float*)d_in[5];
    const float* in_proj_w  = (const float*)d_in[6];
    const float* conv1d_w   = (const float*)d_in[7];
    const float* conv1d_b   = (const float*)d_in[8];
    const float* x_proj_w   = (const float*)d_in[9];
    const float* dt_proj_w  = (const float*)d_in[10];
    const float* dt_proj_b  = (const float*)d_in[11];
    const float* A_log      = (const float*)d_in[12];
    const float* Dvec       = (const float*)d_in[13];
    const float* out_proj_w = (const float*)d_in[14];
    const float* ln_w       = (const float*)d_in[15];
    const float* ln_b       = (const float*)d_in[16];
    const float* head_w     = (const float*)d_in[17];
    float* out = (float*)d_out;

    // ---- workspace carving (256B aligned) ----
    char* base = (char*)d_ws;
    size_t off = 0;
    auto carve = [&](size_t bytes) -> void* {
        void* p = base + off;
        off += (bytes + 255) & ~(size_t)255;
        return p;
    };
    unsigned short* inproj_bf  = (unsigned short*)carve((size_t)(2 * DINNER) * DMODEL * 2);
    unsigned short* xproj_bf   = (unsigned short*)carve((size_t)96 * DINNER * 2);
    unsigned short* dtproj_bf  = (unsigned short*)carve((size_t)DINNER * DTRANK * 2);
    unsigned short* outproj_bf = (unsigned short*)carve((size_t)DMODEL * DINNER * 2);
    unsigned short* head_bf    = (unsigned short*)carve((size_t)C_OUT * DMODEL * 2);
    float*          Aneg       = (float*)carve((size_t)DINNER * DSTATE * 4);
    unsigned short* cur_bf     = (unsigned short*)carve((size_t)MROWS * DMODEL * 2);
    float*          xz         = (float*)carve((size_t)MROWS * 2 * DINNER * 4);
    float*          xact       = (float*)carve((size_t)MROWS * DINNER * 4);
    unsigned short* xact_bf    = (unsigned short*)carve((size_t)MROWS * DINNER * 2);
    float*          xdbc       = (float*)carve((size_t)MROWS * 96 * 4);
    unsigned short* xdbc_bf    = (unsigned short*)carve((size_t)MROWS * 96 * 2);
    float*          dtpre      = (float*)carve((size_t)MROWS * DINNER * 4);
    float*          dtbuf      = (float*)carve((size_t)MROWS * DINNER * 4);
    unsigned short* yg_bf      = (unsigned short*)carve((size_t)MROWS * DINNER * 2);
    float*          mo         = (float*)carve((size_t)MROWS * DMODEL * 4);
    unsigned short* cat_bf     = (unsigned short*)carve((size_t)CATROWS * DMODEL * 2);

    auto cvt = [&](const float* src, unsigned short* dst, int n) {
        f32_to_bf16_kernel<<<(n + 255) / 256, 256, 0, stream>>>(src, dst, n);
    };

    // ---- weight prep (bf16) + A = -exp(A_log) ----
    cvt(in_proj_w,  inproj_bf,  2 * DINNER * DMODEL);
    cvt(x_proj_w,   xproj_bf,   96 * DINNER);
    cvt(dt_proj_w,  dtproj_bf,  DINNER * DTRANK);
    cvt(out_proj_w, outproj_bf, DMODEL * DINNER);
    cvt(head_w,     head_bf,    C_OUT * DMODEL);
    aneg_kernel<<<(DINNER * DSTATE + 255) / 256, 256, 0, stream>>>(A_log, Aneg, DINNER * DSTATE);

    // ---- embedding ----
    embed_kernel<<<(MROWS * DMODEL + 255) / 256, 256, 0, stream>>>(
        x_enc, conv_emb_w, x_mark_enc, temp_w, cur_bf);

    // ---- 2 mamba + LN iterations ----
    for (int it = 0; it < 2; ++it) {
        // xz = cur @ in_proj^T : 8192 x 2048 x 512
        gemm_bf16_wmma<<<dim3((2 * DINNER) / TLN, MROWS / TLM), 256, 0, stream>>>(
            cur_bf, inproj_bf, xz, MROWS, 2 * DINNER, DMODEL, DMODEL, DMODEL, 2 * DINNER);

        conv_silu_kernel<<<(MROWS * DINNER + 255) / 256, 256, 0, stream>>>(
            xz, conv1d_w, conv1d_b, xact, xact_bf);

        // xdbc = x @ x_proj^T : 8192 x 96 x 1024
        gemm_bf16_wmma<<<dim3(1, MROWS / TLM), 256, 0, stream>>>(
            xact_bf, xproj_bf, xdbc, MROWS, 96, DINNER, DINNER, DINNER, 96);

        cvt(xdbc, xdbc_bf, MROWS * 96);

        // dt_pre = xdbc[:, :32] @ dt_proj^T : 8192 x 1024 x 32 (lda = 96)
        gemm_bf16_wmma<<<dim3(DINNER / TLN, MROWS / TLM), 256, 0, stream>>>(
            xdbc_bf, dtproj_bf, dtpre, MROWS, DINNER, DTRANK, 96, DTRANK, DINNER);

        dt_softplus_kernel<<<(MROWS * DINNER + 255) / 256, 256, 0, stream>>>(
            dtpre, dt_proj_b, dtbuf);

        scan_kernel<<<BSZ * (DINNER / 256), 256, 0, stream>>>(
            dtbuf, xact, xdbc, Aneg, Dvec, xz, yg_bf);

        // mo = yg @ out_proj^T : 8192 x 512 x 1024
        gemm_bf16_wmma<<<dim3(DMODEL / TLN, MROWS / TLM), 256, 0, stream>>>(
            yg_bf, outproj_bf, mo, MROWS, DMODEL, DINNER, DINNER, DINNER, DMODEL);

        layernorm_kernel<<<MROWS / 8, 256, 0, stream>>>(
            mo, ln_w, ln_b, cur_bf, cat_bf, it);
    }

    // ---- head: out = cat @ head_w^T : 16384 x 32 x 512 ----
    gemm_bf16_wmma<<<dim3(1, CATROWS / TLM), 256, 0, stream>>>(
        cat_bf, head_bf, out, CATROWS, C_OUT, DMODEL, DMODEL, DMODEL, C_OUT);
}
